// SelfAttentionHead_85727547228737
// MI455X (gfx1250) — matrix operand order repaired and checked by reading, not compiled, not run on hardware
//
#include <hip/hip_runtime.h>

// ---- sizes (compile-time, from the reference) ----
#define B_  8
#define T_  4096
#define C_  512
#define HS_ 64

typedef __bf16 bf16;
typedef __bf16 v16bf __attribute__((ext_vector_type(16)));
typedef float  v8f   __attribute__((ext_vector_type(8)));
typedef unsigned int u32x4 __attribute__((ext_vector_type(4)));
typedef float  f32x4 __attribute__((ext_vector_type(4)));

#define WMMA_BF16(a, b, c) \
    __builtin_amdgcn_wmma_f32_16x16x32_bf16(false, (a), false, (b), (short)0, (c), false, false)

// Load a B-operand fragment: 16 consecutive bf16 starting at p.
// k order used here (same as load_a_frag): N = lane&15, K = 16*(lane>>4) + e.
__device__ __forceinline__ v16bf load_b_frag(const bf16* p) {
    union { v16bf v; u32x4 u[2]; } r;
    r.u[0] = *(const u32x4*)(p);
    r.u[1] = *(const u32x4*)(p + 8);
    return r.v;
}

// Load an A-operand fragment from a row-major bf16 row pointer (row = lane&15).
// Elements 0..15 : K = hi*16 + 0..15.
__device__ __forceinline__ v16bf load_a_frag(const bf16* rowp, int hi) {
    union { v16bf v; u32x4 u[2]; } r;
    r.u[0] = *(const u32x4*)(rowp + hi * 16);
    r.u[1] = *(const u32x4*)(rowp + hi * 16 + 8);
    return r.v;
}

// CDNA5 async copy: 16 bytes global -> LDS, tracked by ASYNCcnt.
// Flat LDS addresses carry the LDS byte offset in addr[31:0] (ISA 10.2),
// so truncating the generic pointer yields the VDST LDS address.
__device__ __forceinline__ void async_copy16(const bf16* g, bf16* l) {
    unsigned           loff = (unsigned)(unsigned long long)(uintptr_t)l;
    unsigned long long ga   = (unsigned long long)(uintptr_t)g;
    asm volatile("global_load_async_to_lds_b128 %0, %1, off"
                 :: "v"(loff), "v"(ga) : "memory");
}

// ---------------------------------------------------------------------------
// Kernel 1: W [C,HS] fp32  ->  W^T [HS,C] bf16  (x3 matrices)
// ---------------------------------------------------------------------------
__global__ __launch_bounds__(256) void prep_w_kernel(
    const float* __restrict__ wk, const float* __restrict__ wq,
    const float* __restrict__ wv,
    bf16* __restrict__ wtq, bf16* __restrict__ wtk, bf16* __restrict__ wtv)
{
    int i = blockIdx.x * blockDim.x + threadIdx.x;
    if (i >= C_ * HS_) return;
    int c = i / HS_;
    int h = i % HS_;
    wtq[h * C_ + c] = (bf16)wq[i];
    wtk[h * C_ + c] = (bf16)wk[i];
    wtv[h * C_ + c] = (bf16)wv[i];
}

// ---------------------------------------------------------------------------
// Kernel 2: projections. One wave computes one 16-row tile of Q, K, V.
//   Q,K stored row-major bf16 [B,T,HS]; V stored transposed bf16 [B,HS,T].
// ---------------------------------------------------------------------------
__global__ __launch_bounds__(256) void proj_kernel(
    const float* __restrict__ x,
    const bf16* __restrict__ wtq, const bf16* __restrict__ wtk,
    const bf16* __restrict__ wtv,
    bf16* __restrict__ qb, bf16* __restrict__ kb, bf16* __restrict__ vt)
{
    const int wave = (blockIdx.x * blockDim.x + threadIdx.x) >> 5;
    const int lane = threadIdx.x & 31;
    const int lo = lane & 15, hi = lane >> 4;
    const int tilesPerB = T_ / 16;
    const int b  = wave / tilesPerB;
    const int t0 = (wave % tilesPerB) * 16;

    const float* xrow = x + ((size_t)b * T_ + t0 + lo) * C_;
    const bf16* wts[3] = { wtq, wtk, wtv };

    v8f acc[3][4];
    const v8f vzero = {0.f, 0.f, 0.f, 0.f, 0.f, 0.f, 0.f, 0.f};
#pragma unroll
    for (int m = 0; m < 3; ++m)
#pragma unroll
        for (int n = 0; n < 4; ++n) acc[m][n] = vzero;

    for (int kc = 0; kc < C_ / 32; ++kc) {
        // A fragment: x tile fp32 -> bf16
        f32x4 f0 = *(const f32x4*)(xrow + kc * 32 + hi * 16);
        f32x4 f1 = *(const f32x4*)(xrow + kc * 32 + hi * 16 + 4);
        f32x4 f2 = *(const f32x4*)(xrow + kc * 32 + hi * 16 + 8);
        f32x4 f3 = *(const f32x4*)(xrow + kc * 32 + hi * 16 + 12);
        v16bf a;
#pragma unroll
        for (int i = 0; i < 4; ++i) {
            a[i]      = (bf16)f0[i];
            a[4 + i]  = (bf16)f1[i];
            a[8 + i]  = (bf16)f2[i];
            a[12 + i] = (bf16)f3[i];
        }
#pragma unroll
        for (int m = 0; m < 3; ++m) {
#pragma unroll
            for (int n = 0; n < 4; ++n) {
                v16bf bfrag = load_b_frag(
                    wts[m] + (size_t)(n * 16 + lo) * C_ + kc * 32 + hi * 16);
                acc[m][n] = WMMA_BF16(a, bfrag, acc[m][n]);
            }
        }
    }

    // D layout: row = v + hi*8, col = n*16 + lo
#pragma unroll
    for (int n = 0; n < 4; ++n) {
#pragma unroll
        for (int v = 0; v < 8; ++v) {
            int row = t0 + v + hi * 8;
            int col = n * 16 + lo;
            qb[((size_t)b * T_ + row) * HS_ + col] = (bf16)acc[0][n][v];
            kb[((size_t)b * T_ + row) * HS_ + col] = (bf16)acc[1][n][v];
            vt[((size_t)b * HS_ + col) * T_ + row] = (bf16)acc[2][n][v];
        }
    }
}

// ---------------------------------------------------------------------------
// Cooperative staging of one 32-key chunk: K rows [s0..s0+31][0..63] (4 KB,
// contiguous in global) and V^T slice [0..63][s0..s0+31] (64 rows x 64 B).
// 128 threads x 4 async 16B transfers.
// ---------------------------------------------------------------------------
__device__ __forceinline__ void stage_chunk(const bf16* kb_b, const bf16* vt_b,
                                            int s0, bf16* kdst, bf16* vdst,
                                            int tid)
{
    const bf16* kg = kb_b + (size_t)s0 * HS_;
#pragma unroll
    for (int i = 0; i < 2; ++i) {
        int off = (tid + i * 128) * 8;                 // bf16 units, 16 B each
        async_copy16(kg + off, kdst + off);
    }
#pragma unroll
    for (int i = 0; i < 2; ++i) {
        int t = tid + i * 128;
        int h = t >> 2, part = (t & 3) * 8;
        async_copy16(vt_b + (size_t)h * T_ + s0 + part, vdst + h * 32 + part);
    }
}

// ---------------------------------------------------------------------------
// Kernel 3: causal flash attention. 4 waves/block = 4 consecutive 16-query
// tiles sharing double-buffered async-staged K/V chunks in LDS.
// ---------------------------------------------------------------------------
__global__ __launch_bounds__(128) void attn_kernel(
    const bf16* __restrict__ qb, const bf16* __restrict__ kb,
    const bf16* __restrict__ vt, float* __restrict__ out)
{
    __shared__ bf16 ldsK[2][32][64];   // K chunk, row-major [s][h]   (2 x 4 KB)
    __shared__ bf16 ldsV[2][64][32];   // V^T chunk, row-major [h][s] (2 x 4 KB)
    __shared__ bf16 plds[4][16][32];   // per-wave P scratch (D -> A transpose)

    const int tid  = threadIdx.x;
    const int wib  = tid >> 5;
    const int lane = tid & 31;
    const int lo = lane & 15, hi = lane >> 4;
    const int tilesPerB = T_ / 16;
    const int qtg = blockIdx.x * 4 + wib;
    const int b   = qtg / tilesPerB;
    const int q0  = (qtg % tilesPerB) * 16;
    const int q0b = ((blockIdx.x * 4) % tilesPerB) * 16;   // block's first tile
    const int nmax = (q0b + 95) >> 5;   // chunks covering s <= q0b+63

    const float scale = 0.04419417382415922f;   // C^-0.5 = 1/sqrt(512)

    const bf16* kb_b = kb + (size_t)b * T_ * HS_;
    const bf16* vt_b = vt + (size_t)b * HS_ * T_;

    // Q tile in A layout (held in registers for the whole pass)
    const bf16* qrow = qb + ((size_t)b * T_ + q0 + lo) * HS_;
    v16bf aq0 = load_a_frag(qrow, hi);        // h = 0..31
    v16bf aq1 = load_a_frag(qrow + 32, hi);   // h = 32..63

    const v8f vzero = {0.f, 0.f, 0.f, 0.f, 0.f, 0.f, 0.f, 0.f};
    v8f o[4];
#pragma unroll
    for (int n = 0; n < 4; ++n) o[n] = vzero;
    float mrow[8], lrow[8];
#pragma unroll
    for (int v = 0; v < 8; ++v) { mrow[v] = -3.0e38f; lrow[v] = 0.f; }

    bf16* myp = &plds[wib][0][0];

    // prologue: stage chunk 0 into buffer 0
    stage_chunk(kb_b, vt_b, 0, &ldsK[0][0][0], &ldsV[0][0][0], tid);

    for (int sc = 0; sc < nmax; ++sc) {
        const int cur = sc & 1;
        const int s0  = sc * 32;

        // my async loads for buf[cur] done; barrier -> everyone's done, and
        // all waves have finished reading buf[1-cur] (last read at sc-1).
        asm volatile("s_wait_asynccnt 0x0" ::: "memory");
        __syncthreads();

        if (sc + 1 < nmax)
            stage_chunk(kb_b, vt_b, s0 + 32,
                        &ldsK[1 - cur][0][0], &ldsV[1 - cur][0][0], tid);

        const bf16* kbuf = &ldsK[cur][0][0];
        const bf16* vbuf = &ldsV[cur][0][0];

        // ---- scores: S[16q x 32s] = Q (16x64) * K^T (64x32), from LDS ----
        v8f sacc[2] = { vzero, vzero };
#pragma unroll
        for (int j = 0; j < 2; ++j) {
            const bf16* krow = kbuf + (j * 16 + lo) * 64 + hi * 16;
            v16bf bk0 = load_b_frag(krow);        // h = hi*16 .. +15
            v16bf bk1 = load_b_frag(krow + 32);   // h = 32 + hi*16 .. +15
            sacc[j] = WMMA_BF16(aq0, bk0, sacc[j]);
            sacc[j] = WMMA_BF16(aq1, bk1, sacc[j]);
        }

        // ---- scale + causal mask + online softmax (per D-layout row) ----
#pragma unroll
        for (int v = 0; v < 8; ++v) {
            const int row = q0 + v + hi * 8;
            float x0 = sacc[0][v] * scale;
            float x1 = sacc[1][v] * scale;
            if (s0 + lo > row)      x0 = -1.0e9f;
            if (s0 + 16 + lo > row) x1 = -1.0e9f;

            float r = fmaxf(x0, x1);
#pragma unroll
            for (int off = 8; off; off >>= 1)
                r = fmaxf(r, __shfl_xor(r, off, 32));   // 16-lane half reduce
            const float nm = fmaxf(mrow[v], r);
            const float al = __expf(mrow[v] - nm);
            const float p0 = __expf(x0 - nm);
            const float p1 = __expf(x1 - nm);
            float rs = p0 + p1;
#pragma unroll
            for (int off = 8; off; off >>= 1)
                rs += __shfl_xor(rs, off, 32);
            lrow[v] = lrow[v] * al + rs;
            mrow[v] = nm;
#pragma unroll
            for (int n = 0; n < 4; ++n) o[n][v] *= al;   // rescale O rows

            myp[(v + hi * 8) * 32 + lo]      = (bf16)p0;
            myp[(v + hi * 8) * 32 + 16 + lo] = (bf16)p1;
        }

        asm volatile("s_wait_dscnt 0x0" ::: "memory");

        // P back as A fragment (row = lo, K = s within chunk)
        union { v16bf v; u32x4 u[2]; } pa;
        pa.u[0] = *(const u32x4*)(myp + lo * 32 + hi * 16);
        pa.u[1] = *(const u32x4*)(myp + lo * 32 + hi * 16 + 8);

        // ---- O += P (16x32) * V (32x64), V^T fragments from LDS ----
#pragma unroll
        for (int n = 0; n < 4; ++n) {
            v16bf bv = load_b_frag(vbuf + (n * 16 + lo) * 32 + hi * 16);
            o[n] = WMMA_BF16(pa.v, bv, o[n]);
        }
    }

    // ---- epilogue: out = O / l ----
#pragma unroll
    for (int n = 0; n < 4; ++n) {
#pragma unroll
        for (int v = 0; v < 8; ++v) {
            const int row = q0 + v + hi * 8;
            out[((size_t)b * T_ + row) * HS_ + n * 16 + lo] = o[n][v] / lrow[v];
        }
    }
}

// ---------------------------------------------------------------------------
extern "C" void kernel_launch(void* const* d_in, const int* in_sizes, int n_in,
                              void* d_out, int out_size, void* d_ws,
                              size_t ws_size, hipStream_t stream)
{
    const float* x  = (const float*)d_in[0];
    // d_in[1] = mask (tril) -- causality is applied analytically in-kernel
    const float* Wk = (const float*)d_in[2];
    const float* Wq = (const float*)d_in[3];
    const float* Wv = (const float*)d_in[4];
    float* out = (float*)d_out;

    char* ws = (char*)d_ws;
    const size_t szQKV = (size_t)B_ * T_ * HS_ * sizeof(bf16);   // 4 MB each
    bf16* qb  = (bf16*)(ws);
    bf16* kb  = (bf16*)(ws + szQKV);
    bf16* vt  = (bf16*)(ws + 2 * szQKV);
    bf16* wtq = (bf16*)(ws + 3 * szQKV);
    bf16* wtk = wtq + C_ * HS_;
    bf16* wtv = wtk + C_ * HS_;

    prep_w_kernel<<<(C_ * HS_ + 255) / 256, 256, 0, stream>>>(
        Wk, Wq, Wv, wtq, wtk, wtv);

    // one wave per 16-row tile, 8 waves / block
    proj_kernel<<<(B_ * T_ / 16) / 8, 256, 0, stream>>>(
        x, wtq, wtk, wtv, qb, kb, vt);

    // one wave per 16-query tile, 4 waves / block (shared K/V staging)
    attn_kernel<<<(B_ * T_ / 16) / 4, 128, 0, stream>>>(qb, kb, vt, out);
}
